// LSTM_23364622090912
// MI455X (gfx1250) — compile-verified, matrix operand order checked by reading, not run on hardware
//
#include <hip/hip_runtime.h>
#include <hip/hip_bf16.h>

// ---------------- problem constants ----------------
#define BB   512            // batch
#define SS   1024           // seq len
#define HH   2048           // hidden
#define OO   128            // output
#define NZ   8192           // 4*H gate columns
#define KK   2048           // reduction dim (h)
#define PITCH 260           // LDS row pitch in floats (256 + 4 pad -> conflict-free)

typedef __attribute__((ext_vector_type(16))) __bf16 v16bf;
typedef __attribute__((ext_vector_type(8)))  __bf16 v8bf;
typedef __attribute__((ext_vector_type(8)))  float  v8f;

// ---------------- helpers ----------------
__device__ __forceinline__ unsigned short f2bf(float f) {
    union { float f; unsigned u; } v; v.f = f;
    unsigned u = v.u;
    unsigned r = u + 0x7FFFu + ((u >> 16) & 1u);   // round-to-nearest-even
    return (unsigned short)(r >> 16);
}

__device__ __forceinline__ float bflo(unsigned u) {
    union { unsigned u; float f; } v; v.u = u << 16; return v.f;
}
__device__ __forceinline__ float bfhi(unsigned u) {
    union { unsigned u; float f; } v; v.u = u & 0xffff0000u; return v.f;
}

__device__ __forceinline__ float sigm(float x) {
    return 1.0f / (1.0f + __expf(-x));
}

__device__ __forceinline__ float tanh_fast(float x) {
    float t = __expf(-2.0f * fabsf(x));
    float r = (1.0f - t) / (1.0f + t);
    return copysignf(r, x);
}

// Load one 16-lane-striped bf16 WMMA fragment: per lane, two contiguous
// 16-byte runs of K values (K = +0..7 and K = +16..23 relative to the lane's
// K base). Emits two global_load_b128 per lane.
__device__ __forceinline__ v16bf load_frag(const unsigned short* __restrict__ p) {
    v8bf lo = *reinterpret_cast<const v8bf*>(p);
    v8bf hi = *reinterpret_cast<const v8bf*>(p + 16);
    v16bf r;
#pragma unroll
    for (int e = 0; e < 8; ++e) { r[e] = lo[e]; r[e + 8] = hi[e]; }
    return r;
}

// ---------------- weight preprocessing ----------------
// Wq is [H, 1+H] fp32 row-major (col 0 = x weight). Build:
//   Wh  [NZ, KK] bf16 row-major (gate-major rows), wx [NZ], bias [NZ] fp32.
__global__ void lstm_prep(const float* __restrict__ Wg, const float* __restrict__ Wi,
                          const float* __restrict__ Wf, const float* __restrict__ Wo,
                          const float* __restrict__ bg, const float* __restrict__ bi,
                          const float* __restrict__ bf_, const float* __restrict__ bo,
                          unsigned short* __restrict__ Wh,
                          float* __restrict__ wx, float* __restrict__ bias) {
    const long total = (long)NZ * KK;
    for (long idx = (long)blockIdx.x * blockDim.x + threadIdx.x; idx < total;
         idx += (long)gridDim.x * blockDim.x) {
        int j = (int)(idx / KK);
        int k = (int)(idx % KK);
        int q = j >> 11;            // gate
        int r = j & (HH - 1);       // row within gate
        const float* Wq = (q == 0) ? Wg : (q == 1) ? Wi : (q == 2) ? Wf : Wo;
        Wh[idx] = f2bf(Wq[(long)r * (HH + 1) + 1 + k]);
        if (k == 0) {
            wx[j] = Wq[(long)r * (HH + 1)];
            const float* bq = (q == 0) ? bg : (q == 1) ? bi : (q == 2) ? bf_ : bo;
            bias[j] = bq[r];
        }
    }
}

// ---------------- state init (h0 = c0 = 0) ----------------
__global__ void lstm_init(unsigned short* __restrict__ hb, float* __restrict__ c) {
    int idx = blockIdx.x * blockDim.x + threadIdx.x;   // BB*HH threads
    hb[idx] = 0;
    c[idx]  = 0.0f;
}

// ---------------- fused LSTM step ------------------------------------------
// One kernel per timestep:  z = h @ Wh^T  (WMMA, bf16 in / f32 acc),
// staged through LDS, then in-block gate nonlinearities + c/h update.
//
// Block: 256 threads = 8 waves in a 2(M) x 4(gate) grid.
// Block tile: 128 batch rows x 64 hidden units (x 4 gates = 256 z columns).
// Wave tile: 64x64 -> 16 v8f accumulators, 16 WMMAs per K-chunk of 32.
// Grid: (HH/64, BB/128) = (32, 4) = 128 blocks.
// h is double-buffered across steps (hin/hout) to avoid the cross-block
// RAW race created by fusing the state update into the GEMM kernel.
__global__ __launch_bounds__(256) void lstm_step(
        const unsigned short* __restrict__ hin,   // [BB, KK] bf16 (step t)
        unsigned short* __restrict__ hout,        // [BB, KK] bf16 (step t+1)
        const unsigned short* __restrict__ Wh,    // [NZ, KK] bf16
        const float* __restrict__ x,              // [BB, SS]
        const float* __restrict__ wx,             // [NZ]
        const float* __restrict__ bias,           // [NZ]
        float* __restrict__ c,                    // [BB, HH] fp32 (in place)
        int t) {
    extern __shared__ float zt[];                 // [128][PITCH]

    const int lane = threadIdx.x & 31;
    const int w    = threadIdx.x >> 5;
    const int l16  = lane & 15;
    const int lh   = lane >> 4;          // K-half selector
    const int wm   = w & 1;              // M half within block
    const int q    = w >> 1;             // gate 0..3

    const int Am0 = blockIdx.y * 128;    // batch-row base of block
    const int j0  = blockIdx.x * 64;     // hidden-unit base of block
    const int Am  = Am0 + wm * 64;
    const int Wrow = q * HH + j0;        // row base into Wh for this gate

    v8f acc[4][4] = {};

    const unsigned short* aBase = hin + (size_t)(Am + l16) * KK + lh * 8;
    const unsigned short* bBase = Wh  + (size_t)(Wrow + l16) * KK + lh * 8;

    for (int k0 = 0; k0 < KK; k0 += 32) {
        v16bf a[4], b[4];
#pragma unroll
        for (int i = 0; i < 4; ++i)
            a[i] = load_frag(aBase + (size_t)(i * 16) * KK + k0);
#pragma unroll
        for (int j = 0; j < 4; ++j)
            b[j] = load_frag(bBase + (size_t)(j * 16) * KK + k0);
#pragma unroll
        for (int i = 0; i < 4; ++i)
#pragma unroll
            for (int j = 0; j < 4; ++j)
                acc[i][j] = __builtin_amdgcn_wmma_f32_16x16x32_bf16(
                    false, a[i], false, b[j], (short)0, acc[i][j], false, false);
    }

    // Stage accumulators to LDS. C/D layout: lane<16 -> M=v, N=lane ;
    // lane>=16 -> M=v+8, N=lane-16. LDS col = gate*64 + local unit.
#pragma unroll
    for (int i = 0; i < 4; ++i) {
#pragma unroll
        for (int j = 0; j < 4; ++j) {
            float* p = zt + (size_t)(wm * 64 + i * 16 + lh * 8) * PITCH
                          + q * 64 + j * 16 + l16;
#pragma unroll
            for (int v = 0; v < 8; ++v) { *p = acc[i][j][v]; p += PITCH; }
        }
    }
    __syncthreads();

    // Pointwise gate update: 128 rows x 64 units = 8192 elems, 32 per thread.
    const int tid = threadIdx.x;
#pragma unroll 4
    for (int it = 0; it < 32; ++it) {
        const int e  = tid + it * 256;
        const int rl = e >> 6;           // local row 0..127
        const int jl = e & 63;           // local hidden unit
        const int brow = Am0 + rl;
        const int jh   = j0 + jl;
        const float xt = x[(size_t)brow * SS + t];
        const float* zr = zt + (size_t)rl * PITCH;

        float zg = zr[jl]        + bias[jh]          + xt * wx[jh];
        float zi = zr[64  + jl]  + bias[jh + HH]     + xt * wx[jh + HH];
        float zf = zr[128 + jl]  + bias[jh + 2*HH]   + xt * wx[jh + 2*HH];
        float zo = zr[192 + jl]  + bias[jh + 3*HH]   + xt * wx[jh + 3*HH];

        float g  = tanh_fast(zg);
        float ig = sigm(zi);
        float fg = sigm(zf);
        float og = sigm(zo);

        const size_t idx = (size_t)brow * HH + jh;
        float cv = g * ig + c[idx] * fg;
        float hv = tanh_fast(cv) * og;

        c[idx]    = cv;
        hout[idx] = f2bf(hv);
    }
}

// ---------------- final output GEMM: out = h_last @ Wy^T + by ----------------
__global__ void lstm_out(const unsigned short* __restrict__ hb,  // [BB, HH] bf16
                         const float* __restrict__ Wy,           // [OO, HH]
                         const float* __restrict__ by,           // [OO]
                         float* __restrict__ out) {              // [BB, OO]
    const int b = blockIdx.x;        // 512 blocks
    const int o = threadIdx.x;       // 128 threads
    const uint4*  hr = reinterpret_cast<const uint4*>(hb + (size_t)b * HH);
    const float*  wr = Wy + (size_t)o * HH;
    float acc = 0.0f;
#pragma unroll 2
    for (int k = 0; k < HH / 8; ++k) {
        uint4 hv = hr[k];                       // 8 bf16
        const float* wk = wr + k * 8;
        acc += bflo(hv.x) * wk[0] + bfhi(hv.x) * wk[1];
        acc += bflo(hv.y) * wk[2] + bfhi(hv.y) * wk[3];
        acc += bflo(hv.z) * wk[4] + bfhi(hv.z) * wk[5];
        acc += bflo(hv.w) * wk[6] + bfhi(hv.w) * wk[7];
    }
    out[(size_t)b * OO + o] = acc + by[o];
}

// ---------------- host driver ----------------
extern "C" void kernel_launch(void* const* d_in, const int* in_sizes, int n_in,
                              void* d_out, int out_size, void* d_ws, size_t ws_size,
                              hipStream_t stream) {
    const float* x  = (const float*)d_in[0];
    const float* Wg = (const float*)d_in[1];
    const float* bg = (const float*)d_in[2];
    const float* Wi = (const float*)d_in[3];
    const float* bi = (const float*)d_in[4];
    const float* Wf = (const float*)d_in[5];
    const float* bf_ = (const float*)d_in[6];
    const float* Wo = (const float*)d_in[7];
    const float* bo = (const float*)d_in[8];
    const float* Wy = (const float*)d_in[9];
    const float* by = (const float*)d_in[10];
    float* out = (float*)d_out;

    // workspace carve-up (~42 MB total)
    char* ws = (char*)d_ws;
    size_t off = 0;
    unsigned short* Wh   = (unsigned short*)(ws + off); off += (size_t)NZ * KK * 2;  // 33.5 MB
    float*          wx   = (float*)(ws + off);          off += (size_t)NZ * 4;       // 32 KB
    float*          bias = (float*)(ws + off);          off += (size_t)NZ * 4;       // 32 KB
    unsigned short* hb0  = (unsigned short*)(ws + off); off += (size_t)BB * HH * 2;  // 2 MB
    unsigned short* hb1  = (unsigned short*)(ws + off); off += (size_t)BB * HH * 2;  // 2 MB
    float*          c    = (float*)(ws + off);          off += (size_t)BB * HH * 4;  // 4 MB

    // 1) pack weights -> bf16 (every call; deterministic, no caching)
    lstm_prep<<<8192, 256, 0, stream>>>(Wg, Wi, Wf, Wo, bg, bi, bf_, bo, Wh, wx, bias);

    // 2) zero initial state (h0 buffer + c)
    lstm_init<<<(BB * HH) / 256, 256, 0, stream>>>(hb0, c);

    // 3) time loop: one fused WMMA-GEMM + gate-update kernel per step,
    //    h double-buffered (SS even -> final h lands in hb0)
    const dim3 ggrid(HH / 64, BB / 128);
    const size_t ldsBytes = (size_t)128 * PITCH * sizeof(float);   // 133,120 B
    for (int t = 0; t < SS; ++t) {
        const unsigned short* hin = (t & 1) ? hb1 : hb0;
        unsigned short*      hoot = (t & 1) ? hb0 : hb1;
        lstm_step<<<ggrid, 256, ldsBytes, stream>>>(hin, hoot, Wh, x, wx, bias, c, t);
    }

    // 4) output projection from final bf16 h
    lstm_out<<<BB, OO, 0, stream>>>(hb0, Wy, by, out);
}
